// GNN_EDGE_15496242004280
// MI455X (gfx1250) — matrix operand order repaired and checked by reading, not compiled, not run on hardware
//
#include <hip/hip_runtime.h>
#include <hip/hip_bf16.h>

// ---------------------------------------------------------------------------
// GNN edge-layer, CDNA5 (gfx1250) implementation.
// B=16, N=64, C=512.  All GEMMs via v_wmma_f32_16x16x32_bf16 (wave32),
// 2-stage ping-pong K pipeline with load-defined (copy-free) fragments.
// ---------------------------------------------------------------------------

typedef __attribute__((ext_vector_type(16))) __bf16 v16bf;
typedef __attribute__((ext_vector_type(8)))  float  v8f;

#define CB   512          // channel dim
#define NN   64           // nodes
#define BB   16           // batch
#define MROW_NODE (BB*NN)         // 1024
#define MROW_EDGE (BB*NN*NN)      // 65536

struct U4x2 { uint4 a, b; };

static __device__ __forceinline__ unsigned short f32_to_bf16(float f) {
    unsigned int u = __builtin_bit_cast(unsigned int, f);
    unsigned int lsb = (u >> 16) & 1u;
    u += 0x7fffu + lsb;                 // round to nearest even
    return (unsigned short)(u >> 16);
}

// A fragment: 16x32 bf16, row-major source [row][k], ld elements.
// Lane L: row = L%16, group g = L/16.
// VGPR v: k = (v>=4 ? 16 : 0) + g*8 + (v%4)*2  -> two 16B loads per lane.
static __device__ __forceinline__ v16bf load_fragA(const unsigned short* base,
                                                   int ld, int k0, int lane) {
    int r = lane & 15, g = lane >> 4;
    const unsigned short* p = base + (size_t)r * ld + k0;
    U4x2 t;
    t.a = *(const uint4*)(p + g * 8);
    t.b = *(const uint4*)(p + 16 + g * 8);
    return __builtin_bit_cast(v16bf, t);
}

// B fragment: 32x16 bf16 with B[k,n] = W[n,k] (W row-major [n][k]).
// Lane L: n = L%16, group g = L/16.
// VGPR v: k = (v>=4 ? 16 : 0) + g*8 + (v%4)*2  -> two 16B loads per lane.
static __device__ __forceinline__ v16bf load_fragB(const unsigned short* wbase,
                                                   int ld, int k0, int lane) {
    int n = lane & 15, g = lane >> 4;
    const unsigned short* p = wbase + (size_t)n * ld + k0;
    U4x2 t;
    t.a = *(const uint4*)(p + g * 8);
    t.b = *(const uint4*)(p + 16 + g * 8);
    return __builtin_bit_cast(v16bf, t);
}

#define WMMA_BF16(A, Bf, Cacc) \
    __builtin_amdgcn_wmma_f32_16x16x32_bf16(false, (A), false, (Bf), (short)0, (Cacc), false, false)

// ---------------------------------------------------------------------------
// 16x512 @ 512x512 row-strip GEMM core, 2-stage ping-pong over K.
// One wave covers 128 output columns (8 n-tiles); acc[8] of v8f.
// Every fragment is defined by a load (never a reg-reg copy), so the
// compiler keeps two physical fragment sets and alternates between them;
// each 18-load batch is covered by the 8-deep WMMA chain of the other set.
// ---------------------------------------------------------------------------
static __device__ __forceinline__ void wmma_row_gemm(
    const unsigned short* __restrict__ Abase,
    const unsigned short* __restrict__ W,
    int ncol0, int lane, v8f acc[8]) {
    const unsigned short* wb[8];
#pragma unroll
    for (int t = 0; t < 8; t++) wb[t] = W + (size_t)(ncol0 + t * 16) * CB;

    v16bf a0 = load_fragA(Abase, CB, 0, lane);
    v16bf b0[8];
#pragma unroll
    for (int t = 0; t < 8; t++) b0[t] = load_fragB(wb[t], CB, 0, lane);

#pragma unroll 1
    for (int k0 = 0; k0 < CB - 64; k0 += 64) {
        // stage 1 loads (k0+32)
        v16bf a1 = load_fragA(Abase, CB, k0 + 32, lane);
        v16bf b1[8];
#pragma unroll
        for (int t = 0; t < 8; t++) b1[t] = load_fragB(wb[t], CB, k0 + 32, lane);
        __builtin_prefetch(Abase + (size_t)(lane & 15) * CB + k0 + 128, 0, 3);
        // consume stage 0 (k0)
#pragma unroll
        for (int t = 0; t < 8; t++) acc[t] = WMMA_BF16(a0, b0[t], acc[t]);
        // stage 0 loads (k0+64)
        a0 = load_fragA(Abase, CB, k0 + 64, lane);
#pragma unroll
        for (int t = 0; t < 8; t++) b0[t] = load_fragB(wb[t], CB, k0 + 64, lane);
        // consume stage 1 (k0+32)
#pragma unroll
        for (int t = 0; t < 8; t++) acc[t] = WMMA_BF16(a1, b1[t], acc[t]);
    }
    // tail: (a0,b0) hold K=448; load and consume K=480
    {
        v16bf a1 = load_fragA(Abase, CB, CB - 32, lane);
        v16bf b1[8];
#pragma unroll
        for (int t = 0; t < 8; t++) b1[t] = load_fragB(wb[t], CB, CB - 32, lane);
#pragma unroll
        for (int t = 0; t < 8; t++) acc[t] = WMMA_BF16(a0, b0[t], acc[t]);
#pragma unroll
        for (int t = 0; t < 8; t++) acc[t] = WMMA_BF16(a1, b1[t], acc[t]);
    }
}

// ---------------------------------------------------------------------------
// fp32 -> bf16 conversion, vectorized float4 -> ushort4 (grid-stride)
// ---------------------------------------------------------------------------
__global__ void k_f32_to_bf16(const float* __restrict__ in,
                              unsigned short* __restrict__ out, long n4) {
    long i = (long)blockIdx.x * blockDim.x + threadIdx.x;
    long stride = (long)gridDim.x * blockDim.x;
    const float4* in4 = (const float4*)in;
    ushort4* out4 = (ushort4*)out;
    for (; i < n4; i += stride) {
        float4 v = in4[i];
        ushort4 o;
        o.x = f32_to_bf16(v.x); o.y = f32_to_bf16(v.y);
        o.z = f32_to_bf16(v.z); o.w = f32_to_bf16(v.w);
        out4[i] = o;
    }
}

// edge prep: fp32 copy into d_out region (working fp32 edge) + bf16 operand
__global__ void k_edge_prep(const float* __restrict__ in,
                            unsigned short* __restrict__ bf,
                            float* __restrict__ f32copy, long n4) {
    long i = (long)blockIdx.x * blockDim.x + threadIdx.x;
    long stride = (long)gridDim.x * blockDim.x;
    const float4* in4 = (const float4*)in;
    float4* cp4 = (float4*)f32copy;
    ushort4* bf4 = (ushort4*)bf;
    for (; i < n4; i += stride) {
        float4 v = in4[i];
        cp4[i] = v;
        ushort4 o;
        o.x = f32_to_bf16(v.x); o.y = f32_to_bf16(v.y);
        o.z = f32_to_bf16(v.z); o.w = f32_to_bf16(v.w);
        bf4[i] = o;
    }
}

// ---------------------------------------------------------------------------
// Node GEMM: out = x(1024x512) @ W.T(512x512), 4 weights on blockIdx.y.
// Tile: 16 rows x 512 cols per workgroup; 4 waves x 8 n-tiles.
// ---------------------------------------------------------------------------
__global__ __launch_bounds__(128)
void k_node_gemm(const unsigned short* __restrict__ A,
                 const unsigned short* __restrict__ w0,
                 const unsigned short* __restrict__ w1,
                 const unsigned short* __restrict__ w2,
                 const unsigned short* __restrict__ w3,
                 float* __restrict__ o0, float* __restrict__ o1,
                 float* __restrict__ o2, float* __restrict__ o3) {
    const unsigned short* W;
    float* O;
    switch (blockIdx.y) {
        case 0:  W = w0; O = o0; break;
        case 1:  W = w1; O = o1; break;
        case 2:  W = w2; O = o2; break;
        default: W = w3; O = o3; break;
    }
    const int row0 = blockIdx.x * 16;
    const int lane = threadIdx.x & 31;
    const int wave = threadIdx.x >> 5;
    const int ncol0 = wave * 128;

    v8f acc[8] = {};
    wmma_row_gemm(A + (size_t)row0 * CB, W, ncol0, lane, acc);

    // D layout: lane L holds col = ntile*16 + L%16, rows (L/16)*8 + r
    const int g = lane >> 4, cl = lane & 15;
#pragma unroll
    for (int t = 0; t < 8; t++) {
        int col = ncol0 + t * 16 + cl;
#pragma unroll
        for (int r = 0; r < 8; r++) {
            O[(size_t)(row0 + g * 8 + r) * CB + col] = acc[t][r];
        }
    }
}

// ---------------------------------------------------------------------------
// Edge fused: e_lin = edge_bf16 @ WE.T, m = e_lin + Vix[i] + Vjx[j],
// edge += relu(LN(m));  writes edge fp32 (in place) and bf16 (next layer).
// One workgroup per 16-row strip (fixed b,i; j0..j0+15) x full 512 cols.
// ---------------------------------------------------------------------------
__global__ __launch_bounds__(128)
void k_edge_fused(unsigned short* __restrict__ edge_bf,
                  const unsigned short* __restrict__ WE,
                  const float* __restrict__ Vix,
                  const float* __restrict__ Vjx,
                  const float* __restrict__ ge,
                  const float* __restrict__ be,
                  float* __restrict__ edge_f32) {
    __shared__ float rs[16];
    __shared__ float rq[16];

    const int strip = blockIdx.x;            // 0..4095
    const long row0 = (long)strip * 16;
    const int b  = strip >> 8;               // 256 strips per batch
    const int rem = (strip & 255) * 16;      // i*64 + j0
    const int i  = rem >> 6;
    const int j0 = rem & 63;

    const int lane = threadIdx.x & 31;
    const int wave = threadIdx.x >> 5;
    const int ncol0 = wave * 128;

    if (threadIdx.x < 16) { rs[threadIdx.x] = 0.f; rq[threadIdx.x] = 0.f; }
    __syncthreads();

    v8f acc[8] = {};
    wmma_row_gemm(edge_bf + (size_t)row0 * CB, WE, ncol0, lane, acc);

    const int g = lane >> 4, cl = lane & 15;
    float vix[8];
#pragma unroll
    for (int t = 0; t < 8; t++)
        vix[t] = Vix[(size_t)(b * NN + i) * CB + ncol0 + t * 16 + cl];

    // m = e_lin + Vix + Vjx; accumulate LN stats per row via LDS atomics
#pragma unroll
    for (int r = 0; r < 8; r++) {
        const int m = g * 8 + r;
        const int j = j0 + m;
        float s = 0.f, q = 0.f;
#pragma unroll
        for (int t = 0; t < 8; t++) {
            float v = acc[t][r] + vix[t] +
                      Vjx[(size_t)(b * NN + j) * CB + ncol0 + t * 16 + cl];
            acc[t][r] = v;
            s += v;
            q += v * v;
        }
        atomicAdd(&rs[m], s);
        atomicAdd(&rq[m], q);
    }
    __syncthreads();

#pragma unroll
    for (int r = 0; r < 8; r++) {
        const int m = g * 8 + r;
        const float mean = rs[m] * (1.f / CB);
        const float var  = rq[m] * (1.f / CB) - mean * mean;
        const float rstd = rsqrtf(var + 1e-5f);
#pragma unroll
        for (int t = 0; t < 8; t++) {
            const int col = ncol0 + t * 16 + cl;
            float y = (acc[t][r] - mean) * rstd * ge[col] + be[col];
            y = fmaxf(y, 0.f);
            const size_t idx = (size_t)(row0 + m) * CB + col;
            const float en = edge_f32[idx] + y;
            edge_f32[idx] = en;
            edge_bf[idx]  = f32_to_bf16(en);
        }
    }
}

// ---------------------------------------------------------------------------
// Node update: e = softmax_j(sigmoid(edge)), agg = (e . Ujx)/64,
// x = relu(res + LN(xU + agg)).  One workgroup per (b,i), 256 thr x 2 chans.
// ---------------------------------------------------------------------------
__global__ __launch_bounds__(256)
void k_node_update(const float* __restrict__ edge_f32,
                   const float* __restrict__ Ujx,
                   const float* __restrict__ xU,
                   const float* __restrict__ xres,
                   const float* __restrict__ gv,
                   const float* __restrict__ bv,
                   float* __restrict__ x_out,
                   unsigned short* __restrict__ x_bf_out) {
    __shared__ float rs[256];
    __shared__ float rq[256];
    const int bi = blockIdx.x;           // b*64 + i
    const int b  = bi >> 6;
    const int tid = threadIdx.x;

    float xn[2];
#pragma unroll
    for (int h = 0; h < 2; h++) {
        const int c = tid + h * 256;
        const float* erow = edge_f32 + (size_t)bi * NN * CB + c;
        float denom = 0.f, num = 0.f;
        for (int j = 0; j < NN; j++) {
            const float v = erow[(size_t)j * CB];
            const float s = 1.f / (1.f + __expf(-v));   // sigmoid
            const float w = __expf(s);                  // softmax weight (s in (0,1))
            denom += w;
            num   += w * Ujx[(size_t)(b * NN + j) * CB + c];
        }
        const float agg = (num / denom) * (1.f / NN);
        xn[h] = xU[(size_t)bi * CB + c] + agg;
    }

    rs[tid] = xn[0] + xn[1];
    rq[tid] = xn[0] * xn[0] + xn[1] * xn[1];
    __syncthreads();
    for (int s = 128; s > 0; s >>= 1) {
        if (tid < s) { rs[tid] += rs[tid + s]; rq[tid] += rq[tid + s]; }
        __syncthreads();
    }
    const float mean = rs[0] * (1.f / CB);
    const float var  = rq[0] * (1.f / CB) - mean * mean;
    const float rstd = rsqrtf(var + 1e-5f);

#pragma unroll
    for (int h = 0; h < 2; h++) {
        const int c = tid + h * 256;
        const float y = (xn[h] - mean) * rstd * gv[c] + bv[c];
        const float o = fmaxf(xres[(size_t)bi * CB + c] + y, 0.f);
        x_out[(size_t)bi * CB + c]   = o;
        x_bf_out[(size_t)bi * CB + c] = f32_to_bf16(o);
    }
}

// ---------------------------------------------------------------------------
// Host-side orchestration
// ---------------------------------------------------------------------------
extern "C" void kernel_launch(void* const* d_in, const int* in_sizes, int n_in,
                              void* d_out, int out_size, void* d_ws, size_t ws_size,
                              hipStream_t stream) {
    const float* x_in    = (const float*)d_in[0];
    const float* edge_in = (const float*)d_in[1];
    const float* Wf[10];
    for (int i = 0; i < 10; i++) Wf[i] = (const float*)d_in[2 + i];
    // dict order: ge1, gv1, ge2, gv2, be1, bv1, be2, bv2
    const float* ge1 = (const float*)d_in[12];
    const float* gv1 = (const float*)d_in[13];
    const float* ge2 = (const float*)d_in[14];
    const float* gv2 = (const float*)d_in[15];
    const float* be1 = (const float*)d_in[16];
    const float* bv1 = (const float*)d_in[17];
    const float* be2 = (const float*)d_in[18];
    const float* bv2 = (const float*)d_in[19];

    // Workspace carve (all chunks 16B-aligned)
    char* p = (char*)d_ws;
    unsigned short* x_bf = (unsigned short*)p;   p += (size_t)MROW_NODE * CB * 2;
    unsigned short* edge_bf = (unsigned short*)p; p += (size_t)MROW_EDGE * CB * 2;
    unsigned short* w_bf = (unsigned short*)p;   p += (size_t)10 * CB * CB * 2;
    float* Vix = (float*)p;  p += (size_t)MROW_NODE * CB * 4;
    float* Vjx = (float*)p;  p += (size_t)MROW_NODE * CB * 4;
    float* Ujx = (float*)p;  p += (size_t)MROW_NODE * CB * 4;
    float* xU  = (float*)p;  p += (size_t)MROW_NODE * CB * 4;
    float* x_mid = (float*)p;

    float* x_out    = (float*)d_out;                       // (B,N,C)
    float* edge_f32 = x_out + (size_t)MROW_NODE * CB;      // (B,N*N,C) working+final

    const long n_node = (long)MROW_NODE * CB;   // 524288
    const long n_edge = (long)MROW_EDGE * CB;   // 33554432

    // Conversions (vectorized x4)
    k_f32_to_bf16<<<512, 256, 0, stream>>>(x_in, x_bf, n_node / 4);
    k_edge_prep<<<8192, 256, 0, stream>>>(edge_in, edge_bf, edge_f32, n_edge / 4);
    for (int i = 0; i < 10; i++)
        k_f32_to_bf16<<<256, 256, 0, stream>>>(Wf[i], w_bf + (size_t)i * CB * CB,
                                               (long)(CB * CB) / 4);
    auto wslot = [&](int s) { return w_bf + (size_t)s * CB * CB; };
    // slots: WA1=0 WB1=1 WE1=2 WU1=3 WV1=4 | WA2=5 WB2=6 WE2=7 WU2=8 WV2=9

    // ---- Layer 1 ----
    k_node_gemm<<<dim3(MROW_NODE / 16, 4), 128, 0, stream>>>(
        x_bf, wslot(0), wslot(1), wslot(4), wslot(3), Vix, Vjx, Ujx, xU);
    k_edge_fused<<<MROW_EDGE / 16, 128, 0, stream>>>(
        edge_bf, wslot(2), Vix, Vjx, ge1, be1, edge_f32);
    k_node_update<<<MROW_NODE, 256, 0, stream>>>(
        edge_f32, Ujx, xU, x_in, gv1, bv1, x_mid, x_bf);

    // ---- Layer 2 ----
    k_node_gemm<<<dim3(MROW_NODE / 16, 4), 128, 0, stream>>>(
        x_bf, wslot(5), wslot(6), wslot(9), wslot(8), Vix, Vjx, Ujx, xU);
    k_edge_fused<<<MROW_EDGE / 16, 128, 0, stream>>>(
        edge_bf, wslot(7), Vix, Vjx, ge2, be2, edge_f32);
    k_node_update<<<MROW_NODE, 256, 0, stream>>>(
        edge_f32, Ujx, xU, x_mid, gv2, bv2, x_out, x_bf);
}
